// Transformer_32719060861061
// MI455X (gfx1250) — matrix operand order repaired and checked by reading, not compiled, not run on hardware
//
#include <hip/hip_runtime.h>
#include <hip/hip_bf16.h>

// ---------------- model constants ----------------
#define B_SZ    2
#define T_SEQ   1024
#define V_VOC   2048
#define D_MODEL 1024
#define N_HEAD  16
#define N_LAYER 8
#define DK_HEAD 64
#define M_ROWS  (B_SZ * T_SEQ)   // 2048
#define QLD     (3 * D_MODEL)    // fused QKV row stride (3072)

typedef unsigned short ushort_t;
typedef __attribute__((ext_vector_type(16))) __bf16 v16bf;
typedef __attribute__((ext_vector_type(8)))  float  v8f;

union UV {
    uint4          q4[2];
    unsigned short s[16];
    v16bf          v;
};
union U8 {
    uint4          q;
    unsigned short s[8];
};

#define WMMA_BF16(a, b, c) \
    __builtin_amdgcn_wmma_f32_16x16x32_bf16(false, (a), false, (b), (short)0, (c), false, false)

__device__ __forceinline__ unsigned short f2bf(float f) {
    unsigned int u = __builtin_bit_cast(unsigned int, f);
    unsigned int r = u + 0x7FFFu + ((u >> 16) & 1u);   // round-to-nearest-even
    return (unsigned short)(r >> 16);
}

// ---- CDNA5 async global->LDS copy (ASYNCcnt path, portable via inline asm) ----
__device__ __forceinline__ void async_cp16(void* lds, const void* gsrc) {
    unsigned loff = (unsigned)(size_t)lds;   // low 32 bits of LDS aperture addr = LDS offset
    asm volatile("global_load_async_to_lds_b128 %0, %1, off"
                 :: "v"(loff), "v"(gsrc) : "memory");
}
__device__ __forceinline__ void wait_async0() {
    asm volatile("s_wait_asynccnt 0" ::: "memory");
}

#define MODE_BIAS 0
#define MODE_POS  1
#define MODE_RELU 2
#define MODE_RES  3

// =====================================================================
// GEMM: C[M,N] = act(A[M,K](bf16) * Bw[N,K](bf16, pre-transposed) + bias)
// 128x128 tile, BK=32, 256 thr = 8 wave32, each wave 64x32 -> 8 WMMA/step.
// Double-buffered LDS fed by async global->LDS b128 copies.
// =====================================================================
#define PITCH 40   // halfwords per LDS row (80B: 16B aligned, conflict-spread)

__global__ void __launch_bounds__(256)
gemm_bf16(const ushort_t* __restrict__ A, const ushort_t* __restrict__ Bw,
          const float* __restrict__ bias, const float* __restrict__ extra,
          void* __restrict__ Cout, int M, int N, int Kd, int mode, int obf)
{
    __shared__ unsigned short sA[2][128 * PITCH];   // [row][k]
    __shared__ unsigned short sB[2][128 * PITCH];   // [n][k]

    const int tid  = threadIdx.x;
    const int lane = tid & 31;
    const int wave = tid >> 5;
    const int wr   = wave >> 2;          // 0..1  -> 64-row strip
    const int wc   = wave & 3;           // 0..3  -> 32-col strip
    const int hf   = lane >> 4;
    const int l16  = lane & 15;

    const int rowBase = blockIdx.y * 128;
    const int colBase = blockIdx.x * 128;

    // staging assignment: thread -> (row r, 16-elem chunk h16) of each tile
    const int sr  = tid >> 1;
    const int h16 = (tid & 1) * 16;
    const ushort_t* gA = A  + (size_t)(rowBase + sr) * Kd + h16;
    const ushort_t* gB = Bw + (size_t)(colBase + sr) * Kd + h16;
    const unsigned sAoff = sr * PITCH + h16;
    const unsigned sBoff = sr * PITCH + h16;

    v8f acc[4][2];
    const v8f vzero = {0.f, 0.f, 0.f, 0.f, 0.f, 0.f, 0.f, 0.f};
    for (int i = 0; i < 4; ++i)
        for (int j = 0; j < 2; ++j) acc[i][j] = vzero;

    const int nk = Kd >> 5;

    // preload tile 0 into buffer 0
    async_cp16(&sA[0][sAoff], gA);
    async_cp16(&sA[0][sAoff + 8], gA + 8);
    async_cp16(&sB[0][sBoff], gB);
    async_cp16(&sB[0][sBoff + 8], gB + 8);

    for (int ks = 0; ks < nk; ++ks) {
        const int cur = ks & 1;
        wait_async0();
        __syncthreads();
        if (ks + 1 < nk) {               // async-stage next tile while computing
            const int nxt = cur ^ 1;
            const int k1  = (ks + 1) << 5;
            async_cp16(&sA[nxt][sAoff], gA + k1);
            async_cp16(&sA[nxt][sAoff + 8], gA + k1 + 8);
            async_cp16(&sB[nxt][sBoff], gB + k1);
            async_cp16(&sB[nxt][sBoff + 8], gB + k1 + 8);
        }
        UV af[4], bfq[2];
        for (int i = 0; i < 4; ++i) {
            int r  = wr * 64 + i * 16 + l16;
            int k1 = hf * 8;                       // lane<16: K0..7/16..23; lane>=16: K8..15/24..31
            const uint4* p1 = (const uint4*)&sA[cur][r * PITCH + k1];
            const uint4* p2 = (const uint4*)&sA[cur][r * PITCH + k1 + 16];
            af[i].q4[0] = p1[0];
            af[i].q4[1] = p2[0];
        }
        for (int j = 0; j < 2; ++j) {
            int n  = wc * 32 + j * 16 + l16;
            int kb = hf * 8;                       // lane<16: K0..7/16..23; lane>=16: K8..15/24..31
            const uint4* p = (const uint4*)&sB[cur][n * PITCH + kb];
            bfq[j].q4[0] = p[0];
            bfq[j].q4[1] = p[2];
        }
        for (int i = 0; i < 4; ++i)
            for (int j = 0; j < 2; ++j)
                acc[i][j] = WMMA_BF16(af[i].v, bfq[j].v, acc[i][j]);
    }

    // ---- epilogue ----
    for (int i = 0; i < 4; ++i)
        for (int j = 0; j < 2; ++j)
            for (int v = 0; v < 8; ++v) {
                int r = rowBase + wr * 64 + i * 16 + hf * 8 + v;
                int c = colBase + wc * 32 + j * 16 + l16;
                float val = acc[i][j][v] + bias[c];
                if (mode == MODE_POS)       val += extra[(size_t)(r % T_SEQ) * N + c];
                else if (mode == MODE_RELU) val  = fmaxf(val, 0.f);
                else if (mode == MODE_RES)  val += extra[(size_t)r * N + c];
                if (obf) ((ushort_t*)Cout)[(size_t)r * N + c] = f2bf(val);
                else     ((float*)Cout)[(size_t)r * N + c]    = val;
            }
}

// =====================================================================
// Flash attention on fused bf16 QKV [M, 3*D]: block = (b, h, 64 Q rows),
// 4 waves x 16 rows. Online softmax, causal; WMMA bf16; O -> bf16.
// =====================================================================
__global__ void __launch_bounds__(128)
attn_kernel(const ushort_t* __restrict__ QKV, ushort_t* __restrict__ O)
{
    __shared__ unsigned short sKT[64 * 72];  // [s][dk]
    __shared__ unsigned short sVT[64 * 72];  // [dk][s]
    __shared__ unsigned short sP [64 * 72];  // [row][s]

    const int tid  = threadIdx.x;
    const int lane = tid & 31;
    const int wave = tid >> 5;
    const int hf   = lane >> 4;
    const int l16  = lane & 15;

    const int qbase = blockIdx.x * 64;
    const int h     = blockIdx.y;
    const int b     = blockIdx.z;
    const int hcol  = h * DK_HEAD;

    // ---- Q A-fragments straight from global bf16 ----
    UV qf[2];
    {
        const int trow = qbase + wave * 16 + l16;
        const ushort_t* qp = QKV + (size_t)(b * T_SEQ + trow) * QLD + hcol;
        for (int s = 0; s < 2; ++s) {
            int k1 = s * 32 + hf * 8;
            qf[s].q4[0] = *(const uint4*)(qp + k1);
            qf[s].q4[1] = *(const uint4*)(qp + k1 + 16);
        }
    }

    float mrow[8], lrow[8];
    v8f oacc[4];
    const v8f vzero = {0.f, 0.f, 0.f, 0.f, 0.f, 0.f, 0.f, 0.f};
    for (int v = 0; v < 8; ++v) { mrow[v] = -1e30f; lrow[v] = 0.f; }
    for (int j = 0; j < 4; ++j) oacc[j] = vzero;

    const int smax = qbase + 63;
    for (int sb = 0; sb <= smax; sb += 64) {
        // ---- K tile: async b128 copies; V tile: transpose via b16 stores ----
        for (int i = 0; i < 4; ++i) {               // 512 uint4 chunks / 128 thr
            int c    = tid + i * 128;
            int srow = c >> 3, ch = c & 7;
            const ushort_t* krow = QKV + (size_t)(b * T_SEQ + sb + srow) * QLD
                                       + D_MODEL + hcol + ch * 8;
            async_cp16(&sKT[srow * 72 + ch * 8], krow);
            U8 vv;
            vv.q = *(const uint4*)(QKV + (size_t)(b * T_SEQ + sb + srow) * QLD
                                       + 2 * D_MODEL + hcol + ch * 8);
            for (int e = 0; e < 8; ++e)
                sVT[(ch * 8 + e) * 72 + srow] = vv.s[e];
        }
        wait_async0();
        __syncthreads();

        // ---- S strip = Q K^T (16x64 / wave) ----
        float sv[4][8];
        for (int j = 0; j < 4; ++j) {
            v8f sacc = vzero;
            for (int s = 0; s < 2; ++s) {
                UV kf;
                int n  = j * 16 + l16;
                int kb = s * 32 + hf * 8;
                const uint4* p = (const uint4*)&sKT[n * 72 + kb];
                kf.q4[0] = p[0];
                kf.q4[1] = p[2];
                sacc = WMMA_BF16(qf[s].v, kf.v, sacc);
            }
            for (int v = 0; v < 8; ++v) {
                int scol = sb + j * 16 + l16;
                int tr   = qbase + wave * 16 + hf * 8 + v;
                sv[j][v] = (scol <= tr) ? sacc[v] * 0.125f : -1e30f;  // scale + causal mask
            }
        }

        // ---- online softmax (16-lane butterfly per C row group) ----
        float fsc[8];
        for (int v = 0; v < 8; ++v) {
            float m = fmaxf(fmaxf(sv[0][v], sv[1][v]), fmaxf(sv[2][v], sv[3][v]));
            for (int d = 8; d >= 1; d >>= 1) m = fmaxf(m, __shfl_xor(m, d, 32));
            float mn = fmaxf(mrow[v], m);
            fsc[v]   = __expf(mrow[v] - mn);
            mrow[v]  = mn;
            float ls = 0.f;
            for (int j = 0; j < 4; ++j) { sv[j][v] = __expf(sv[j][v] - mn); ls += sv[j][v]; }
            for (int d = 8; d >= 1; d >>= 1) ls += __shfl_xor(ls, d, 32);
            lrow[v] = lrow[v] * fsc[v] + ls;
        }
        for (int j = 0; j < 4; ++j)
            for (int v = 0; v < 8; ++v) oacc[j][v] *= fsc[v];

        // ---- P -> LDS bf16 (A-fragment layout source) ----
        for (int j = 0; j < 4; ++j)
            for (int v = 0; v < 8; ++v)
                sP[(wave * 16 + hf * 8 + v) * 72 + j * 16 + l16] = f2bf(sv[j][v]);
        __syncthreads();

        // ---- O += P V ----
        UV pf[2];
        for (int s = 0; s < 2; ++s) {
            int k1 = s * 32 + hf * 8;
            pf[s].q4[0] = *(const uint4*)&sP[(wave * 16 + l16) * 72 + k1];
            pf[s].q4[1] = *(const uint4*)&sP[(wave * 16 + l16) * 72 + k1 + 16];
        }
        for (int j = 0; j < 4; ++j) {
            for (int s = 0; s < 2; ++s) {
                UV vf;
                int n  = j * 16 + l16;
                int kb = s * 32 + hf * 8;
                const uint4* p = (const uint4*)&sVT[n * 72 + kb];
                vf.q4[0] = p[0];
                vf.q4[1] = p[2];
                oacc[j] = WMMA_BF16(pf[s].v, vf.v, oacc[j]);
            }
        }
        __syncthreads();
    }

    // ---- normalize, write heads-concatenated bf16 output [M, D] ----
    for (int v = 0; v < 8; ++v) {
        int   tr  = qbase + wave * 16 + hf * 8 + v;
        float inv = 1.f / lrow[v];
        for (int j = 0; j < 4; ++j)
            O[(size_t)(b * T_SEQ + tr) * D_MODEL + hcol + j * 16 + l16] =
                f2bf(oacc[j][v] * inv);
    }
}

// =====================================================================
// LayerNorm row kernel: f32 in -> bf16 out (GEMM A operand).
// =====================================================================
__global__ void __launch_bounds__(256)
ln_kernel(const float* __restrict__ X, const float* __restrict__ g,
          const float* __restrict__ bta, ushort_t* __restrict__ Y)
{
    __shared__ float s1[256], s2[256];
    const int row = blockIdx.x, tid = threadIdx.x;
    const float* xp = X + (size_t)row * D_MODEL;
    float vals[4], sum = 0.f, sq = 0.f;
    for (int i = 0; i < 4; ++i) {
        float x = xp[tid + i * 256];
        vals[i] = x; sum += x; sq += x * x;
    }
    s1[tid] = sum; s2[tid] = sq;
    __syncthreads();
    for (int st = 128; st > 0; st >>= 1) {
        if (tid < st) { s1[tid] += s1[tid + st]; s2[tid] += s2[tid + st]; }
        __syncthreads();
    }
    float mean = s1[0] * (1.f / D_MODEL);
    float var  = s2[0] * (1.f / D_MODEL) - mean * mean;
    float rstd = rsqrtf(var + 1e-5f);
    ushort_t* yp = Y + (size_t)row * D_MODEL;
    for (int i = 0; i < 4; ++i) {
        int c = tid + i * 256;
        yp[c] = f2bf((vals[i] - mean) * rstd * g[c] + bta[c]);
    }
}

// =====================================================================
// Weight prep: transpose+convert f32 [K,N] (or headed [H,D,DK]) -> bf16 [N,K].
// 32x32 LDS tile keeps both global sides coalesced.
// =====================================================================
__global__ void __launch_bounds__(256)
prep_w(const float* __restrict__ W, ushort_t* __restrict__ out,
       int Kd, int N, int headed)
{
    __shared__ float t[32][33];
    const int nb = blockIdx.x * 32, kb = blockIdx.y * 32;
    const int tx = threadIdx.x & 31, ty = threadIdx.x >> 5;   // 32 x 8
    for (int i = 0; i < 4; ++i) {
        int k = kb + ty + i * 8, n = nb + tx;
        size_t off = headed
            ? ((size_t)(n >> 6) * Kd * DK_HEAD + (size_t)k * DK_HEAD + (n & 63))
            : ((size_t)k * N + n);
        t[ty + i * 8][tx] = W[off];
    }
    __syncthreads();
    for (int i = 0; i < 4; ++i)
        out[(size_t)(nb + ty + i * 8) * Kd + kb + tx] = f2bf(t[tx][ty + i * 8]);
}

// ---- misc small kernels ----
__global__ void __launch_bounds__(256)
conv_bf16(const float* __restrict__ in, ushort_t* __restrict__ out)
{
    int e = (blockIdx.x * 256 + threadIdx.x) * 4;
    for (int i = 0; i < 4; ++i) out[e + i] = f2bf(in[e + i]);
}

__global__ void __launch_bounds__(256)
biascat3(const float* __restrict__ a, const float* __restrict__ b,
         const float* __restrict__ c, float* __restrict__ o)
{
    int t = blockIdx.x * 256 + threadIdx.x;   // 0..3071
    o[t] = (t < D_MODEL) ? a[t]
         : (t < 2 * D_MODEL) ? b[t - D_MODEL] : c[t - 2 * D_MODEL];
}

// =====================================================================
extern "C" void kernel_launch(void* const* d_in, const int* in_sizes, int n_in,
                              void* d_out, int out_size, void* d_ws, size_t ws_size,
                              hipStream_t stream)
{
    const float* idx     = (const float*)d_in[0];
    const float* tok_w   = (const float*)d_in[1];
    const float* tok_b   = (const float*)d_in[2];
    const float* pos_emb = (const float*)d_in[3];
    const float* ln1_g   = (const float*)d_in[4];
    const float* ln1_b   = (const float*)d_in[5];
    const float* wq      = (const float*)d_in[6];
    const float* wk      = (const float*)d_in[7];
    const float* wv      = (const float*)d_in[8];
    const float* bq      = (const float*)d_in[9];
    const float* bk      = (const float*)d_in[10];
    const float* bv      = (const float*)d_in[11];
    const float* proj_w  = (const float*)d_in[12];
    const float* proj_b  = (const float*)d_in[13];
    const float* ln2_g   = (const float*)d_in[14];
    const float* ln2_b   = (const float*)d_in[15];
    const float* ff1_w   = (const float*)d_in[16];
    const float* ff1_b   = (const float*)d_in[17];
    const float* ff2_w   = (const float*)d_in[18];
    const float* ff2_b   = (const float*)d_in[19];
    const float* lnf_g   = (const float*)d_in[20];
    const float* lnf_b   = (const float*)d_in[21];
    const float* head_w  = (const float*)d_in[22];
    const float* head_b  = (const float*)d_in[23];
    (void)in_sizes; (void)n_in; (void)out_size; (void)ws_size;

    const size_t DD = (size_t)D_MODEL * D_MODEL;

    // ---- workspace carve-up (~52 MB) ----
    float*    X     = (float*)d_ws;                          // [M,D] f32 residual
    float*    Bias3 = X + (size_t)M_ROWS * D_MODEL;          // [3072] (+pad)
    ushort_t* Abf   = (ushort_t*)(Bias3 + 4096);             // [M, V] bf16 acts
    ushort_t* Abf2  = Abf  + (size_t)M_ROWS * V_VOC;         // [M, 4D] bf16 acts
    ushort_t* Wbf   = Abf2 + (size_t)M_ROWS * 4 * D_MODEL;   // [4M] bf16 weights (transposed)
    ushort_t* QKVb  = Wbf  + (size_t)4 * DD;                 // [M, 3D] bf16

    const dim3 blk256(256), blk128(128);
    const dim3 gEmb (D_MODEL / 128,     M_ROWS / 128);
    const dim3 gQKV (QLD / 128,         M_ROWS / 128);
    const dim3 gProj(D_MODEL / 128,     M_ROWS / 128);
    const dim3 gFF1 (4 * D_MODEL / 128, M_ROWS / 128);
    const dim3 gHead(V_VOC / 128,       M_ROWS / 128);
    const dim3 gAttn(T_SEQ / 64, N_HEAD, B_SZ);

    // x = idx @ tok_w + tok_b + pos_emb
    conv_bf16<<<(M_ROWS * V_VOC) / 1024, blk256, 0, stream>>>(idx, Abf);
    prep_w<<<dim3(D_MODEL / 32, V_VOC / 32), blk256, 0, stream>>>(tok_w, Wbf, V_VOC, D_MODEL, 0);
    gemm_bf16<<<gEmb, blk256, 0, stream>>>(Abf, Wbf, tok_b, pos_emb, X,
                                           M_ROWS, D_MODEL, V_VOC, MODE_POS, 0);

    for (int i = 0; i < N_LAYER; ++i) {
        // h = LN1(x)  (bf16)
        ln_kernel<<<M_ROWS, blk256, 0, stream>>>(X, ln1_g + i * D_MODEL, ln1_b + i * D_MODEL, Abf);
        // fused QKV: B' = [wq|wk|wv] repacked -> [3072][1024]
        prep_w<<<dim3(D_MODEL / 32, D_MODEL / 32), blk256, 0, stream>>>(wq + i * DD, Wbf,            D_MODEL, D_MODEL, 1);
        prep_w<<<dim3(D_MODEL / 32, D_MODEL / 32), blk256, 0, stream>>>(wk + i * DD, Wbf + DD,       D_MODEL, D_MODEL, 1);
        prep_w<<<dim3(D_MODEL / 32, D_MODEL / 32), blk256, 0, stream>>>(wv + i * DD, Wbf + 2 * DD,   D_MODEL, D_MODEL, 1);
        biascat3<<<QLD / 256, blk256, 0, stream>>>(bq + i * D_MODEL, bk + i * D_MODEL,
                                                   bv + i * D_MODEL, Bias3);
        gemm_bf16<<<gQKV, blk256, 0, stream>>>(Abf, Wbf, Bias3, nullptr, QKVb,
                                               M_ROWS, QLD, D_MODEL, MODE_BIAS, 1);
        // o = causal-softmax(q k^T / 8) v  -> bf16 heads-concat in Abf
        attn_kernel<<<gAttn, blk128, 0, stream>>>(QKVb, Abf);
        // x = x + o @ proj_w + proj_b
        prep_w<<<dim3(D_MODEL / 32, D_MODEL / 32), blk256, 0, stream>>>(proj_w + i * DD, Wbf, D_MODEL, D_MODEL, 0);
        gemm_bf16<<<gProj, blk256, 0, stream>>>(Abf, Wbf, proj_b + i * D_MODEL, X, X,
                                                M_ROWS, D_MODEL, D_MODEL, MODE_RES, 0);
        // x = x + relu(LN2(x) @ ff1 + b1) @ ff2 + b2
        ln_kernel<<<M_ROWS, blk256, 0, stream>>>(X, ln2_g + i * D_MODEL, ln2_b + i * D_MODEL, Abf);
        prep_w<<<dim3(4 * D_MODEL / 32, D_MODEL / 32), blk256, 0, stream>>>(ff1_w + i * 4 * DD, Wbf, D_MODEL, 4 * D_MODEL, 0);
        gemm_bf16<<<gFF1, blk256, 0, stream>>>(Abf, Wbf, ff1_b + i * 4 * D_MODEL, nullptr, Abf2,
                                               M_ROWS, 4 * D_MODEL, D_MODEL, MODE_RELU, 1);
        prep_w<<<dim3(D_MODEL / 32, (4 * D_MODEL) / 32), blk256, 0, stream>>>(ff2_w + i * 4 * DD, Wbf, 4 * D_MODEL, D_MODEL, 0);
        gemm_bf16<<<gProj, blk256, 0, stream>>>(Abf2, Wbf, ff2_b + i * D_MODEL, X, X,
                                                M_ROWS, D_MODEL, 4 * D_MODEL, MODE_RES, 0);
    }

    // logits = LN_f(x) @ head_w + head_b  (f32 out)
    ln_kernel<<<M_ROWS, blk256, 0, stream>>>(X, lnf_g, lnf_b, Abf);
    prep_w<<<dim3(V_VOC / 32, D_MODEL / 32), blk256, 0, stream>>>(head_w, Wbf, D_MODEL, V_VOC, 0);
    gemm_bf16<<<gHead, blk256, 0, stream>>>(Abf, Wbf, head_b, nullptr, d_out,
                                            M_ROWS, V_VOC, D_MODEL, MODE_BIAS, 0);
}
